// RULSTM_85899345920558
// MI455X (gfx1250) — compile-verified, matrix operand order checked by reading, not run on hardware
//
#include <hip/hip_runtime.h>

// ---------------------------------------------------------------------------
// CDNA5 WMMA types (wave32): A/B = 16 halves/lane (8 VGPRs), C/D = 8 f32.
// ---------------------------------------------------------------------------
typedef _Float16 half8 __attribute__((ext_vector_type(8)));
typedef _Float16 v16h  __attribute__((ext_vector_type(16)));
typedef float    v8f   __attribute__((ext_vector_type(8)));

union HalfFrag { v16h v; half8 h[2]; };

// Per-lane base pointer for a 16x32 f16 fragment at (row0, k=0).
// ISA 7.12.2: lane L -> row M = L%16; halves 0..7 = K offset (L>=16?8:0)+0..7,
// halves 8..15 = K offset 16+(L>=16?8:0)+0..7  -> two contiguous 16B loads.
__device__ __forceinline__ const _Float16* frag_base(
    const _Float16* __restrict__ base, int ld, int row0, int lane) {
  return base + (size_t)(row0 + (lane & 15)) * (size_t)ld + ((lane >> 4) << 3);
}

__device__ __forceinline__ v16h frag_load(const _Float16* __restrict__ p, int k0) {
  HalfFrag u;
  u.h[0] = *reinterpret_cast<const half8*>(p + k0);
  u.h[1] = *reinterpret_cast<const half8*>(p + k0 + 16);
  return u.v;
}

__device__ __forceinline__ v8f wmma_f16(v16h a, v16h b, v8f c) {
  return __builtin_amdgcn_wmma_f32_16x16x32_f16(false, a, false, b, (short)0, c,
                                                false, false);
}

// ---------------------------------------------------------------------------
// Generic WMMA GEMM:  D[M,N] = A[M,K] * Bw[N,K]^T (+ bias[n])
// Block 128x128, 8 waves, wave tile 64M x 32N, K unrolled by 64 (2 frag sets).
// REMAP: output row = (m%256)*14 + m/256   ([S,B] -> [B,S] rows, S=14,B=256)
// ---------------------------------------------------------------------------
template <bool HAS_BIAS, bool REMAP>
__global__ __launch_bounds__(256) void gemm_wmma_f16(
    const _Float16* __restrict__ A, int lda,
    const _Float16* __restrict__ Bw, int ldb,
    const float* __restrict__ bias,
    float* __restrict__ D, int N, int K)
{
  const int bm = blockIdx.y << 7;
  const int bn = blockIdx.x << 7;
  const int lane = threadIdx.x & 31;
  const int wid  = threadIdx.x >> 5;
  const int wm = bm + ((wid >> 2) << 6);   // 0 / 64
  const int wn = bn + ((wid & 3) << 5);    // 0,32,64,96
  const int lr = lane & 15;
  const int hi = lane >> 4;

  const _Float16* ap[4];
  const _Float16* bp[2];
#pragma unroll
  for (int i = 0; i < 4; ++i) ap[i] = frag_base(A, lda, wm + (i << 4), lane);
#pragma unroll
  for (int j = 0; j < 2; ++j) bp[j] = frag_base(Bw, ldb, wn + (j << 4), lane);

  v8f zero;
#pragma unroll
  for (int r = 0; r < 8; ++r) zero[r] = 0.0f;
  v8f acc[4][2];
#pragma unroll
  for (int i = 0; i < 4; ++i)
#pragma unroll
    for (int j = 0; j < 2; ++j) acc[i][j] = zero;

  for (int k0 = 0; k0 < K; k0 += 64) {
    v16h a0[4], b0[2], a1[4], b1[2];
#pragma unroll
    for (int i = 0; i < 4; ++i) a0[i] = frag_load(ap[i], k0);
#pragma unroll
    for (int j = 0; j < 2; ++j) b0[j] = frag_load(bp[j], k0);
#pragma unroll
    for (int i = 0; i < 4; ++i) a1[i] = frag_load(ap[i], k0 + 32);
#pragma unroll
    for (int j = 0; j < 2; ++j) b1[j] = frag_load(bp[j], k0 + 32);
#pragma unroll
    for (int i = 0; i < 4; ++i)
#pragma unroll
      for (int j = 0; j < 2; ++j) acc[i][j] = wmma_f16(a0[i], b0[j], acc[i][j]);
#pragma unroll
    for (int i = 0; i < 4; ++i)
#pragma unroll
      for (int j = 0; j < 2; ++j) acc[i][j] = wmma_f16(a1[i], b1[j], acc[i][j]);
  }

#pragma unroll
  for (int i = 0; i < 4; ++i) {
#pragma unroll
    for (int j = 0; j < 2; ++j) {
      const int ncol = wn + (j << 4) + lr;
      if (ncol >= N) continue;
      float bv = 0.0f;
      if (HAS_BIAS) bv = bias[ncol];
#pragma unroll
      for (int r = 0; r < 8; ++r) {
        const int mrow = wm + (i << 4) + r + (hi << 3);
        const size_t orow = REMAP ? (size_t)((mrow & 255) * 14 + (mrow >> 8))
                                  : (size_t)mrow;
        D[orow * (size_t)N + ncol] = acc[i][j][r] + bv;
      }
    }
  }
}

// ---------------------------------------------------------------------------
// Fused recurrent LSTM step:
//   gates(m, g*1024+n) = h16_in[m,:] . Whh16[g*1024+n, :] + xg[m, g*1024+n]
//   c2 = sig(f)*c + sig(i)*tanh(g);  h2 = sig(o)*tanh(c2)
// Block tile: 128 rows x 32 h-columns, computing ALL FOUR gate columns, so
// each lane holds (i,f,g,o) for its (m,n) in 4 accumulators -> fully fused
// gate update, no gates round-trip. h-state ping-pongs (read h16_in, write
// h16_out) to avoid the intra-launch RAW race; c updated in place (tile-owned).
//   step == 0 : rolling mode; archive h2->h16_arch, c2->c_arch (always).
//   step  > 0 : unroll mode; chain t=m/256 skips if step > 15-t and writes
//               h16_arch only at its final step (step == 15-t).
// 8 waves: wave tile 32M x 16N x 4 gates; K unrolled by 64 (2 frag sets).
// ---------------------------------------------------------------------------
__global__ __launch_bounds__(256) void lstm_step_fused(
    const _Float16* __restrict__ h16_in,
    _Float16* __restrict__ h16_out,
    const _Float16* __restrict__ Whh16,   // [4096,1024]
    const float* __restrict__ xg,         // [M,4096]
    float* __restrict__ c,                // [M,1024], in place
    _Float16* __restrict__ h16_arch,
    float* __restrict__ c_arch,           // null in unroll mode
    int step)
{
  const int bn = blockIdx.x << 5;          // 32 h-cols
  const int bm = blockIdx.y << 7;          // 128 rows
  bool final_step = false;
  if (step > 0) {
    const int t = bm >> 8;                 // 256 rows per chain
    if (step > 15 - t) return;
    final_step = (step == 15 - t);
  }
  const int lane = threadIdx.x & 31;
  const int wid  = threadIdx.x >> 5;
  const int wm = bm + ((wid >> 1) << 5);   // 4 m-waves * 32 rows
  const int wn = bn + ((wid & 1) << 4);    // 2 n-waves * 16 cols
  const int lr = lane & 15;
  const int hi = lane >> 4;

  const _Float16* ap[2];
  const _Float16* bp[4];
#pragma unroll
  for (int i = 0; i < 2; ++i) ap[i] = frag_base(h16_in, 1024, wm + (i << 4), lane);
#pragma unroll
  for (int g = 0; g < 4; ++g) bp[g] = frag_base(Whh16, 1024, (g << 10) + wn, lane);

  v8f zero;
#pragma unroll
  for (int r = 0; r < 8; ++r) zero[r] = 0.0f;
  v8f acc[2][4];                           // [m-frag][gate]
#pragma unroll
  for (int i = 0; i < 2; ++i)
#pragma unroll
    for (int g = 0; g < 4; ++g) acc[i][g] = zero;

  for (int k0 = 0; k0 < 1024; k0 += 64) {
    v16h a0[2], b0[4], a1[2], b1[4];
#pragma unroll
    for (int i = 0; i < 2; ++i) a0[i] = frag_load(ap[i], k0);
#pragma unroll
    for (int g = 0; g < 4; ++g) b0[g] = frag_load(bp[g], k0);
#pragma unroll
    for (int i = 0; i < 2; ++i) a1[i] = frag_load(ap[i], k0 + 32);
#pragma unroll
    for (int g = 0; g < 4; ++g) b1[g] = frag_load(bp[g], k0 + 32);
#pragma unroll
    for (int i = 0; i < 2; ++i)
#pragma unroll
      for (int g = 0; g < 4; ++g) acc[i][g] = wmma_f16(a0[i], b0[g], acc[i][g]);
#pragma unroll
    for (int i = 0; i < 2; ++i)
#pragma unroll
      for (int g = 0; g < 4; ++g) acc[i][g] = wmma_f16(a1[i], b1[g], acc[i][g]);
  }

  const bool do_arch = (step == 0) || final_step;
  const int n = wn + lr;
#pragma unroll
  for (int i = 0; i < 2; ++i) {
#pragma unroll
    for (int r = 0; r < 8; ++r) {
      const int m = wm + (i << 4) + r + (hi << 3);
      const size_t b4 = (size_t)m << 12;   // row offset in [M,4096]
      const float gi = acc[i][0][r] + xg[b4 + n];
      const float gf = acc[i][1][r] + xg[b4 + 1024 + n];
      const float gg = acc[i][2][r] + xg[b4 + 2048 + n];
      const float go = acc[i][3][r] + xg[b4 + 3072 + n];
      const float si = 1.0f / (1.0f + __expf(-gi));
      const float sf = 1.0f / (1.0f + __expf(-gf));
      const float so = 1.0f / (1.0f + __expf(-go));
      const size_t idx = ((size_t)m << 10) + n;
      const float c2 = sf * c[idx] + si * tanhf(gg);
      const float h2 = so * tanhf(c2);
      c[idx] = c2;
      h16_out[idx] = (_Float16)h2;
      if (do_arch) h16_arch[idx] = (_Float16)h2;
      if (c_arch) c_arch[idx] = c2;
    }
  }
}

// ---------------------------------------------------------------------------
// Setup kernels
// ---------------------------------------------------------------------------
__global__ void f32_to_f16_pad(const float* __restrict__ src,
                               _Float16* __restrict__ dst,
                               size_t n, size_t npad) {
  const size_t i = (size_t)blockIdx.x * 256 + threadIdx.x;
  if (i < npad) dst[i] = (i < n) ? (_Float16)src[i] : (_Float16)0.0f;
}

// inputs [B=256,S=14,F=1024] -> x16 [S,B,F] fp16
__global__ void make_x16_kernel(const float* __restrict__ inp,
                                _Float16* __restrict__ x16) {
  const size_t i = (size_t)blockIdx.x * 256 + threadIdx.x;
  if (i >= (size_t)3584 * 1024) return;
  const int s = (int)(i >> 18);            // B*F = 2^18
  const int rem = (int)(i & 262143);
  const int b = rem >> 10;
  const int f = rem & 1023;
  x16[i] = (_Float16)inp[((size_t)b * 14 + s) * 1024 + f];
}

__global__ void bias_sum_kernel(const float* __restrict__ a,
                                const float* __restrict__ b,
                                float* __restrict__ o, int n) {
  const int i = blockIdx.x * 256 + threadIdx.x;
  if (i < n) o[i] = a[i] + b[i];
}

__global__ void zero_f32_kernel(float* p, size_t n) {
  const size_t i = (size_t)blockIdx.x * 256 + threadIdx.x;
  if (i < n) p[i] = 0.0f;
}
__global__ void zero_f16_kernel(_Float16* p, size_t n) {
  const size_t i = (size_t)blockIdx.x * 256 + threadIdx.x;
  if (i < n) p[i] = (_Float16)0.0f;
}

// ---------------------------------------------------------------------------
// Host-side orchestration
// ---------------------------------------------------------------------------
extern "C" void kernel_launch(void* const* d_in, const int* in_sizes, int n_in,
                              void* d_out, int out_size, void* d_ws, size_t ws_size,
                              hipStream_t stream) {
  (void)in_sizes; (void)n_in; (void)out_size; (void)ws_size;

  const float* inputs = (const float*)d_in[0];
  const float* Wih_r  = (const float*)d_in[1];
  const float* Whh_r  = (const float*)d_in[2];
  const float* bih_r  = (const float*)d_in[3];
  const float* bhh_r  = (const float*)d_in[4];
  const float* Wih_u  = (const float*)d_in[5];
  const float* Whh_u  = (const float*)d_in[6];
  const float* bih_u  = (const float*)d_in[7];
  const float* bhh_u  = (const float*)d_in[8];
  const float* Wc     = (const float*)d_in[9];
  const float* bc     = (const float*)d_in[10];

  constexpr int Bd = 256, S = 14, F = 1024, H = 1024, C = 2513;
  constexpr int G  = 4 * H;        // 4096
  constexpr int MB = S * Bd;       // 3584
  constexpr int Cpad = 2560;       // C rounded up to 128

  char* ws = (char*)d_ws;
  size_t off = 0;
  auto alloc = [&](size_t bytes) -> void* {
    void* p = ws + off;
    off = (off + bytes + 255) & ~(size_t)255;
    return p;
  };

  _Float16* Wih_r16 = (_Float16*)alloc((size_t)G * F * 2);
  _Float16* Whh_r16 = (_Float16*)alloc((size_t)G * H * 2);
  _Float16* Wih_u16 = (_Float16*)alloc((size_t)G * F * 2);
  _Float16* Whh_u16 = (_Float16*)alloc((size_t)G * H * 2);
  _Float16* Wc16    = (_Float16*)alloc((size_t)Cpad * H * 2);
  _Float16* x16     = (_Float16*)alloc((size_t)MB * F * 2);
  float*    bsum_r  = (float*)alloc((size_t)G * 4);
  float*    bsum_u  = (float*)alloc((size_t)G * 4);
  float*    xg_r    = (float*)alloc((size_t)MB * G * 4);
  float*    xg_u    = (float*)alloc((size_t)MB * G * 4);
  float*    c_roll  = (float*)alloc((size_t)Bd * H * 4);
  _Float16* h16A_r  = (_Float16*)alloc((size_t)Bd * H * 2);
  _Float16* h16B_r  = (_Float16*)alloc((size_t)Bd * H * 2);
  _Float16* hU16A   = (_Float16*)alloc((size_t)MB * H * 2);   // unroll state A
  _Float16* hU16B   = (_Float16*)alloc((size_t)MB * H * 2);   // unroll state B
  float*    cU      = (float*)alloc((size_t)MB * H * 4);      // unroll c state
  _Float16* hFin16  = (_Float16*)alloc((size_t)MB * H * 2);   // final h per chain

  auto cdiv = [](size_t a, size_t b) -> unsigned { return (unsigned)((a + b - 1) / b); };
  const dim3 blk(256);

  // ---- setup: fp16 conversions, transpose, bias sums, zero-init -----------
  f32_to_f16_pad<<<cdiv((size_t)G * F, 256), blk, 0, stream>>>(Wih_r, Wih_r16, (size_t)G * F, (size_t)G * F);
  f32_to_f16_pad<<<cdiv((size_t)G * H, 256), blk, 0, stream>>>(Whh_r, Whh_r16, (size_t)G * H, (size_t)G * H);
  f32_to_f16_pad<<<cdiv((size_t)G * F, 256), blk, 0, stream>>>(Wih_u, Wih_u16, (size_t)G * F, (size_t)G * F);
  f32_to_f16_pad<<<cdiv((size_t)G * H, 256), blk, 0, stream>>>(Whh_u, Whh_u16, (size_t)G * H, (size_t)G * H);
  f32_to_f16_pad<<<cdiv((size_t)Cpad * H, 256), blk, 0, stream>>>(Wc, Wc16, (size_t)C * H, (size_t)Cpad * H);
  make_x16_kernel<<<cdiv((size_t)MB * F, 256), blk, 0, stream>>>(inputs, x16);
  bias_sum_kernel<<<cdiv(G, 256), blk, 0, stream>>>(bih_r, bhh_r, bsum_r, G);
  bias_sum_kernel<<<cdiv(G, 256), blk, 0, stream>>>(bih_u, bhh_u, bsum_u, G);
  zero_f32_kernel<<<cdiv((size_t)Bd * H, 256), blk, 0, stream>>>(c_roll, (size_t)Bd * H);
  zero_f16_kernel<<<cdiv((size_t)Bd * H, 256), blk, 0, stream>>>(h16A_r, (size_t)Bd * H);

  // ---- input projections: xg = x @ Wih^T + (bih + bhh) --------------------
  const dim3 gA(G / 128, MB / 128);
  gemm_wmma_f16<true, false><<<gA, blk, 0, stream>>>(x16, F, Wih_r16, F, bsum_r, xg_r, G, F);
  gemm_wmma_f16<true, false><<<gA, blk, 0, stream>>>(x16, F, Wih_u16, F, bsum_u, xg_u, G, F);

  // ---- rolling LSTM: 14 fused steps; archives h->hU16A (f16), c->cU -------
  const dim3 gR(H / 32, Bd / 128);
  for (int s = 0; s < S; ++s) {
    const _Float16* hin  = (s & 1) ? h16B_r : h16A_r;
    _Float16*       hout = (s & 1) ? h16A_r : h16B_r;
    lstm_step_fused<<<gR, blk, 0, stream>>>(
        hin, hout, Whh_r16, xg_r + (size_t)s * Bd * G, c_roll,
        hU16A + (size_t)s * Bd * H, cU + (size_t)s * Bd * H, 0);
  }

  // ---- unrolling LSTM: 15 fused steps over all 14 chains ------------------
  // chain t (rows t*256..t*256+255) runs steps 1..15-t; at its final step it
  // writes hFin16. State ping-pongs hU16A <-> hU16B; c (cU) updated in place.
  const dim3 gU(H / 32, MB / 128);
  for (int k = 1; k <= S + 1; ++k) {
    const _Float16* hin  = (k & 1) ? hU16A : hU16B;
    _Float16*       hout = (k & 1) ? hU16B : hU16A;
    lstm_step_fused<<<gU, blk, 0, stream>>>(
        hin, hout, Whh_u16, xg_u, cU, hFin16, nullptr, k);
  }

  // ---- classifier: y[b,s,:] = h_n[s,b,:] @ Wc^T + bc ----------------------
  const dim3 gC(cdiv((size_t)C, 128), MB / 128);
  gemm_wmma_f16<true, true><<<gC, blk, 0, stream>>>(hFin16, H, Wc16, H, bc,
                                                    (float*)d_out, C, H);
}
